// ScaledDotProductAttention_77730318123259
// MI455X (gfx1250) — matrix operand order repaired and checked by reading, not compiled, not run on hardware
//
#include <hip/hip_runtime.h>

// ---------------------------------------------------------------------------
// Flash attention (causal) for B=4,H=16,S=2048,DH=128, fp32 in/out.
// Compute: f32 -> f16 WMMA (v_wmma_f32_16x16x32_f16), f32 softmax/accum.
// One workgroup (8 waves) owns a 128-row Q tile; each wave a 16-row stripe.
// K/V tiles staged in LDS as f16 once per workgroup; global loads for tile
// t+1 are software-pipelined behind the compute of tile t. Softmax row
// reductions use v_permlane16_b32 (VALU) instead of ds_bpermute so they
// co-execute with WMMA and stay off the LDS pipe feeding the fragments.
// ---------------------------------------------------------------------------

typedef __attribute__((ext_vector_type(16))) _Float16 v16h;
typedef __attribute__((ext_vector_type(8)))  _Float16 v8h;
typedef __attribute__((ext_vector_type(2)))  _Float16 v2h;
typedef __attribute__((ext_vector_type(8)))  float    v8f;
typedef __attribute__((ext_vector_type(4)))  float    v4f;
typedef __attribute__((ext_vector_type(2)))  float    v2f;

#define B_    4
#define H_    16
#define S_    2048
#define DH_   128
#define QTILE 128      // rows per workgroup (8 waves x 16)
#define KT    32       // keys per inner tile
#define KS_STRIDE 136  // f16 stride for K rows   (272B: 16B-aligned, bank-spread)
#define VT_STRIDE 40   // f16 stride for V^T rows (80B : 16B-aligned, bank-spread)

// ---- VALU lane-xor within each 16-lane row via v_permlane16_b32 ----------
// selector nibble for lane i is (i ^ off); both 16-lane rows use the same
// row-relative selectors, which is exactly our half-wave reduction scope.
__device__ __forceinline__ float permlane16_xor(float v, unsigned s1, unsigned s2) {
#if __has_builtin(__builtin_amdgcn_permlane16)
  int iv = __builtin_bit_cast(int, v);
  int r  = __builtin_amdgcn_permlane16(iv, iv, (int)s1, (int)s2, false, false);
  return __builtin_bit_cast(float, r);
#else
  (void)s1; (void)s2;
  return v; // unreachable fallback path replaced below
#endif
}

__device__ __forceinline__ float rowmax16(float v) {
#if __has_builtin(__builtin_amdgcn_permlane16)
  v = fmaxf(v, permlane16_xor(v, 0x67452301u, 0xEFCDAB89u)); // xor 1
  v = fmaxf(v, permlane16_xor(v, 0x54761032u, 0xDCFE98BAu)); // xor 2
  v = fmaxf(v, permlane16_xor(v, 0x32107654u, 0xBA98FEDCu)); // xor 4
  v = fmaxf(v, permlane16_xor(v, 0xFEDCBA98u, 0x76543210u)); // xor 8
#else
  for (int off = 1; off < 16; off <<= 1) v = fmaxf(v, __shfl_xor(v, off, 32));
#endif
  return v;
}

__device__ __forceinline__ float rowsum16(float v) {
#if __has_builtin(__builtin_amdgcn_permlane16)
  v += permlane16_xor(v, 0x67452301u, 0xEFCDAB89u); // xor 1
  v += permlane16_xor(v, 0x54761032u, 0xDCFE98BAu); // xor 2
  v += permlane16_xor(v, 0x32107654u, 0xBA98FEDCu); // xor 4
  v += permlane16_xor(v, 0xFEDCBA98u, 0x76543210u); // xor 8
#else
  for (int off = 1; off < 16; off <<= 1) v += __shfl_xor(v, off, 32);
#endif
  return v;
}

__launch_bounds__(256, 1)
__global__ void fa_causal_kernel(const float* __restrict__ Km,
                                 const float* __restrict__ Vm,
                                 const float* __restrict__ Qm,
                                 float* __restrict__ Om) {
  __shared__ _Float16 KS[KT * KS_STRIDE];    // K tile, row-major [k][d], f16
  __shared__ _Float16 VT[DH_ * VT_STRIDE];   // V tile, transposed [d][k], f16
  __shared__ _Float16 Pst[8 * 16 * 32];      // per-wave P staging [16][32]

  const int bh    = blockIdx.y;
  const int qbase = blockIdx.x * QTILE;
  const int tid   = threadIdx.x;
  const int wave  = tid >> 5;
  const int lane  = tid & 31;
  const int half  = lane >> 4;   // 0: lanes 0-15, 1: lanes 16-31
  const int l16   = lane & 15;

  const float* Qb = Qm + (size_t)bh * S_ * DH_;
  const float* Kb = Km + (size_t)bh * S_ * DH_;
  const float* Vb = Vm + (size_t)bh * S_ * DH_;
  float*       Ob = Om + (size_t)bh * S_ * DH_;

  const int q0   = qbase + wave * 16;      // first Q row of this wave
  const int qrow = q0 + l16;
  const float scale = 0.08838834764831845f;   // 1/sqrt(128)

  // staging coordinates for this thread (8 float2 elements per tile)
  const int st_k [8] = { (tid      ) >> 6, (tid + 256) >> 6, (tid + 512) >> 6, (tid + 768) >> 6,
                         (tid +1024) >> 6, (tid +1280) >> 6, (tid +1536) >> 6, (tid +1792) >> 6 };
  const int st_dp = tid & 63;              // float2 index within a row (same all i)

  // ---- Q A-fragments: 16x128, 4 chunks of 16x32 f16 (registers, reused) --
  v16h qa[4];
  {
    const float* qr = Qb + (size_t)qrow * DH_;
#pragma unroll
    for (int c = 0; c < 4; ++c) {
      const int d0 = c * 32 + (half ? 8 : 0);
      v4f a0 = *(const v4f*)(qr + d0);
      v4f a1 = *(const v4f*)(qr + d0 + 4);
      v4f a2 = *(const v4f*)(qr + d0 + 16);
      v4f a3 = *(const v4f*)(qr + d0 + 20);
      v16h f;
#pragma unroll
      for (int i = 0; i < 4; ++i) {
        f[i]      = (_Float16)a0[i];
        f[i + 4]  = (_Float16)a1[i];
        f[i + 8]  = (_Float16)a2[i];
        f[i + 12] = (_Float16)a3[i];
      }
      qa[c] = f;
    }
  }

  // ---- accumulators + online-softmax row stats ---------------------------
  v8f acc[8];
#pragma unroll
  for (int n = 0; n < 8; ++n) acc[n] = (v8f){0,0,0,0,0,0,0,0};
  float rowm[8], rowl[8];
#pragma unroll
  for (int r = 0; r < 8; ++r) { rowm[r] = -__builtin_inff(); rowl[r] = 0.f; }

  _Float16* pw = Pst + wave * (16 * 32);

  // ---- software-pipelined tile staging registers (f16, 16 VGPRs total) ---
  v2h kreg[8], vreg[8];
  auto load_tile = [&](int kb) {
#pragma unroll
    for (int i = 0; i < 8; ++i) {
      const size_t off = (size_t)(kb + st_k[i]) * DH_ + st_dp * 2;
      v2f fk = *(const v2f*)(Kb + off);
      v2f fv = *(const v2f*)(Vb + off);
      kreg[i] = (v2h){(_Float16)fk[0], (_Float16)fk[1]};
      vreg[i] = (v2h){(_Float16)fv[0], (_Float16)fv[1]};
    }
  };
  auto store_tile = [&]() {
#pragma unroll
    for (int i = 0; i < 8; ++i) {
      const int k = st_k[i];
      *(v2h*)(KS + k * KS_STRIDE + st_dp * 2) = kreg[i];
      VT[(st_dp * 2    ) * VT_STRIDE + k] = vreg[i][0];
      VT[(st_dp * 2 + 1) * VT_STRIDE + k] = vreg[i][1];
    }
  };

  const int nkt = (qbase + QTILE) / KT;  // causal upper edge for this block
  load_tile(0);

  for (int t = 0; t < nkt; ++t) {
    const int kb = t * KT;

    __syncthreads();                     // previous iteration's readers done
    store_tile();                        // pure ds_store of pre-converted f16
    __syncthreads();

    // issue next tile's global loads now; they overlap compute below
    if (t + 1 < nkt) load_tile(kb + KT);

    if (kb <= q0 + 15) {                 // wave-uniform causal skip
      // ---- scores: two 16-key subtiles, S = Q K^T ------------------------
      v8f s[2];
#pragma unroll
      for (int sub = 0; sub < 2; ++sub) {
        const _Float16* krow =
            KS + (sub * 16 + l16) * KS_STRIDE + (half ? 8 : 0);
        // batch all 8 ds_load_b128 before the wmma chain (pipelined waits)
        v16h kf[4];
#pragma unroll
        for (int c = 0; c < 4; ++c) {
          v8h b0 = *(const v8h*)(krow + c * 32);      // d +0..7  (16B aligned)
          v8h b1 = *(const v8h*)(krow + c * 32 + 16); // d +16..23
          kf[c] = __builtin_shufflevector(b0, b1,
                    0,1,2,3,4,5,6,7,8,9,10,11,12,13,14,15);
        }
        v8f sc = (v8f){0,0,0,0,0,0,0,0};
#pragma unroll
        for (int c = 0; c < 4; ++c)
          sc = __builtin_amdgcn_wmma_f32_16x16x32_f16(
                   false, qa[c], false, kf[c], (short)0, sc, false, false);
        s[sub] = sc;
      }

      // ---- scale (+ causal mask only on diagonal-crossing tiles) ---------
      if (kb + KT - 1 > q0) {            // wave-uniform: tile touches diagonal
#pragma unroll
        for (int sub = 0; sub < 2; ++sub) {
          const int col = kb + sub * 16 + l16;
#pragma unroll
          for (int r = 0; r < 8; ++r) {
            const int row = q0 + r + half * 8;
            const float v = s[sub][r] * scale;
            s[sub][r] = (col > row) ? -__builtin_inff() : v;
          }
        }
      } else {                           // fully visible tile: scale only
#pragma unroll
        for (int sub = 0; sub < 2; ++sub)
#pragma unroll
          for (int r = 0; r < 8; ++r)
            s[sub][r] = s[sub][r] * scale;
      }

      // ---- online softmax: VALU permlane reductions per row --------------
#pragma unroll
      for (int r = 0; r < 8; ++r) {
        const float tm   = rowmax16(fmaxf(s[0][r], s[1][r]));
        const float mnew = fmaxf(rowm[r], tm);
        const float fac  = __expf(rowm[r] - mnew);
        const float p0   = __expf(s[0][r] - mnew);
        const float p1   = __expf(s[1][r] - mnew);
        s[0][r] = p0;
        s[1][r] = p1;
        const float ts = rowsum16(p0 + p1);
        rowl[r] = rowl[r] * fac + ts;
        rowm[r] = mnew;
#pragma unroll
        for (int n = 0; n < 8; ++n) acc[n][r] *= fac;
      }

      // ---- P: C layout -> A layout via per-wave LDS slab -----------------
#pragma unroll
      for (int sub = 0; sub < 2; ++sub)
#pragma unroll
        for (int r = 0; r < 8; ++r) {
          const int row = r + half * 8;
          pw[row * 32 + sub * 16 + l16] = (_Float16)s[sub][r];
        }
      asm volatile("" ::: "memory");     // keep DS store->load order in IR
      v8h plo = *(const v8h*)(pw + l16 * 32 + (half ? 16 : 0));
      v8h phi = *(const v8h*)(pw + l16 * 32 + 8 + (half ? 16 : 0));
      v16h pa = __builtin_shufflevector(plo, phi,
                  0,1,2,3,4,5,6,7,8,9,10,11,12,13,14,15);

      // ---- O += P V : 8 column tiles, B-frags batched in groups of 4 -----
#pragma unroll
      for (int g = 0; g < 2; ++g) {
        v16h vb[4];
#pragma unroll
        for (int j = 0; j < 4; ++j) {
          const int d = (g * 4 + j) * 16 + l16;
          const _Float16* vt = VT + d * VT_STRIDE + (half ? 16 : 0);
          v8h b0 = *(const v8h*)(vt);      // keys +0..7  (16B aligned)
          v8h b1 = *(const v8h*)(vt + 8);  // keys +8..15
          vb[j] = __builtin_shufflevector(b0, b1,
                    0,1,2,3,4,5,6,7,8,9,10,11,12,13,14,15);
        }
#pragma unroll
        for (int j = 0; j < 4; ++j)
          acc[g * 4 + j] = __builtin_amdgcn_wmma_f32_16x16x32_f16(
              false, pa, false, vb[j], (short)0, acc[g * 4 + j], false, false);
      }
    }
  }

  // ---- epilogue: normalize and store (coalesced across lanes 0-15) -------
#pragma unroll
  for (int r = 0; r < 8; ++r) {
    const int row = q0 + r + half * 8;
    const float inv = 1.0f / rowl[r];
#pragma unroll
    for (int n = 0; n < 8; ++n)
      Ob[(size_t)row * DH_ + n * 16 + l16] = acc[n][r] * inv;
  }
}

extern "C" void kernel_launch(void* const* d_in, const int* in_sizes, int n_in,
                              void* d_out, int out_size, void* d_ws, size_t ws_size,
                              hipStream_t stream) {
  // setup_inputs() order: key, value, query (all fp32)
  const float* K = (const float*)d_in[0];
  const float* V = (const float*)d_in[1];
  const float* Q = (const float*)d_in[2];
  float*       O = (float*)d_out;
  dim3 grid(S_ / QTILE, B_ * H_);
  fa_causal_kernel<<<grid, dim3(256), 0, stream>>>(K, V, Q, O);
}
